// SelfAttention_28140625723409
// MI455X (gfx1250) — compile-verified, matrix operand order checked by reading, not run on hardware
//
#include <hip/hip_runtime.h>

// ---------------------------------------------------------------------------
// Self-attention for MI455X (gfx1250): all matmuls via v_wmma_f32_16x16x32_f16
// Q/K/V/attn-out kept in f16 (f32 accumulate), flash-attention inner loop.
// K tiles are brought into LDS by the Tensor Data Mover (tensor_load_to_lds,
// double buffered, s_wait_tensorcnt), with D# LDS padding giving bank-staggered
// 144B rows that line up with the ds_load_b128 fragment gathers.
// ---------------------------------------------------------------------------

typedef __attribute__((ext_vector_type(16))) _Float16 v16h;
typedef __attribute__((ext_vector_type(8)))  _Float16 v8h;
typedef __attribute__((ext_vector_type(4)))  _Float16 v4h;
typedef __attribute__((ext_vector_type(8)))  float    v8f;
typedef __attribute__((ext_vector_type(4)))  float    v4f;
typedef __attribute__((ext_vector_type(4)))  unsigned int u32x4;
typedef __attribute__((ext_vector_type(4)))  int      i32x4;
typedef __attribute__((ext_vector_type(8)))  int      i32x8;

#define D_MODEL 1024
#define N_QKV   3072
#define T_SEQ   2048
#define BATCH   2
#define NH      16
#define HD      64

#define TA_STRIDE 40   // halves per A-tile row (32 data + 8 pad) -> 80B rows
#define TB_STRIDE 40
#define VT_STRIDE 72   // halves per transposed-V row (64 data + 8 pad)
#define P_STRIDE  72
#define KT_STRIDE 72   // K tile row stride in halves (TDM pad: 128B data + 16B pad)
#define KT_SIZE   (64 * KT_STRIDE)

union V16U { v16h v; v8h h[2]; };
__device__ inline v16h make16(v8h lo, v8h hi) { V16U u; u.h[0] = lo; u.h[1] = hi; return u.v; }

__device__ inline v8f wmma_f16(v16h a, v16h b, v8f c) {
    // D = A(16x32 f16) * B(32x16 f16) + C(16x16 f32)
    return __builtin_amdgcn_wmma_f32_16x16x32_f16(
        /*neg_a=*/false, a, /*neg_b=*/false, b,
        /*c_mod=*/(short)0, c, /*reuse_a=*/false, /*reuse_b=*/false);
}

// TDM: DMA a 64x64 f16 row-major tile (row stride HD halves in memory) into
// LDS at lds_off, padding +4 dwords after every 32 dwords (one 128B row) so
// the LDS row stride is 144B = KT_STRIDE halves.
__device__ inline void tdm_load_k_tile(const _Float16* gsrc, unsigned lds_off) {
    unsigned long long ga = (unsigned long long)(size_t)gsrc;
    u32x4 g0;
    g0.x = 1u;                                   // count=1, user mode, no gather
    g0.y = lds_off;                              // LDS byte address
    g0.z = (unsigned)(ga & 0xffffffffu);         // global_addr[31:0]
    g0.w = (unsigned)(ga >> 32) | (2u << 30);    // global_addr[56:32] | type=2
    i32x8 g1;
    g1[0] = (1 << 16)        // data_size = 2 bytes
          | (1 << 20)        // pad_enable
          | (4 << 22)        // pad_interval: 32 dwords (128B)
          | (3 << 25);       // pad_amount: 4 dwords (16B)
    g1[1] = (HD & 0xffff) << 16;          // tensor_dim0[15:0]
    g1[2] = (T_SEQ & 0xffff) << 16;       // tensor_dim0[31:16]=0 | tensor_dim1[15:0]
    g1[3] = (HD & 0xffff) << 16;          // tensor_dim1[31:16]=0 | tile_dim0=64
    g1[4] = 64;                           // tile_dim1=64, tile_dim2=0
    g1[5] = HD;                           // tensor_dim0_stride[31:0]
    g1[6] = 0;
    g1[7] = 0;
    i32x4 g2 = {0, 0, 0, 0};
    i32x4 g3 = {0, 0, 0, 0};
#if __clang_major__ >= 23
    i32x8 g4 = {0, 0, 0, 0, 0, 0, 0, 0};
    __builtin_amdgcn_tensor_load_to_lds(g0, g1, g2, g3, g4, 0);
#else
    __builtin_amdgcn_tensor_load_to_lds(g0, g1, g2, g3, 0);
#endif
}

// ---------------------------------------------------------------------------
// Kernel 1: qkv = x @ w_qkv + b_qkv ; scatter Q/K/V (f16) into [B,H,T,64].
// Q is pre-scaled by 1/sqrt(64). Block tile 128x64, 8 waves, K-step 32.
// ---------------------------------------------------------------------------
__global__ void __launch_bounds__(256)
qkv_gemm_kernel(const float* __restrict__ X, const float* __restrict__ W,
                const float* __restrict__ bias,
                _Float16* __restrict__ Qh, _Float16* __restrict__ Kh,
                _Float16* __restrict__ Vh)
{
    __shared__ __align__(16) _Float16 As[128 * TA_STRIDE];
    __shared__ __align__(16) _Float16 Bs[64 * TB_STRIDE];

    const int tid  = threadIdx.x;
    const int wave = tid >> 5;
    const int lane = tid & 31;
    const int nn   = lane & 15;       // column within 16x16 tile
    const int hs   = lane >> 4;       // half-select (K sub-chunk / row group)
    const int row0 = blockIdx.x * 128;
    const int col0 = blockIdx.y * 64;

    v8f acc[4];
#pragma unroll
    for (int t = 0; t < 4; ++t) acc[t] = {};

    for (int kk = 0; kk < D_MODEL; kk += 32) {
        // Stage A: 128x32 f32 -> f16 in LDS (coalesced float4 loads)
#pragma unroll
        for (int it = 0; it < 4; ++it) {
            int slot = tid + it * 256;
            int r = slot >> 3, cs = slot & 7;
            v4f a = *(const v4f*)(X + (size_t)(row0 + r) * D_MODEL + kk + cs * 4);
            v4h ah = {(_Float16)a.x, (_Float16)a.y, (_Float16)a.z, (_Float16)a.w};
            *(v4h*)(As + r * TA_STRIDE + cs * 4) = ah;
        }
        // Stage B: 32x64 f32 -> f16, column-major in LDS
#pragma unroll
        for (int it = 0; it < 2; ++it) {
            int slot = tid + it * 256;
            int kr = slot >> 4, ns = slot & 15;
            v4f b = *(const v4f*)(W + (size_t)(kk + kr) * N_QKV + col0 + ns * 4);
            Bs[(ns * 4 + 0) * TB_STRIDE + kr] = (_Float16)b.x;
            Bs[(ns * 4 + 1) * TB_STRIDE + kr] = (_Float16)b.y;
            Bs[(ns * 4 + 2) * TB_STRIDE + kr] = (_Float16)b.z;
            Bs[(ns * 4 + 3) * TB_STRIDE + kr] = (_Float16)b.w;
        }
        // Warm GL2 for next K-step while this tile computes
        if (kk + 32 < D_MODEL) {
            __builtin_prefetch(X + (size_t)(row0 + (tid >> 3)) * D_MODEL + kk + 32 + (tid & 7) * 4, 0, 3);
            __builtin_prefetch(W + (size_t)(kk + 32 + (tid >> 4)) * N_QKV + col0 + (tid & 15) * 4, 0, 3);
        }
        __syncthreads();

        const int kb = hs * 8;  // lane's K sub-offsets: kb..kb+7 and kb+16..kb+23
        const _Float16* arow = As + (wave * 16 + nn) * TA_STRIDE;
        v16h afrag = make16(*(const v8h*)(arow + kb), *(const v8h*)(arow + kb + 16));
        v16h bfrag[4];
#pragma unroll
        for (int t = 0; t < 4; ++t) {
            const _Float16* bcol = Bs + (t * 16 + nn) * TB_STRIDE;
            bfrag[t] = make16(*(const v8h*)(bcol + kb), *(const v8h*)(bcol + kb + 16));
        }
#pragma unroll
        for (int t = 0; t < 4; ++t)
            acc[t] = wmma_f16(afrag, bfrag[t], acc[t]);
        __syncthreads();
    }

    // Epilogue: bias, Q-scale, scatter to [B,H,T,64] f16
    const int mbase = row0 + wave * 16 + hs * 8;   // C layout: vgpr i -> row i + hs*8
#pragma unroll
    for (int t = 0; t < 4; ++t) {
        int n     = col0 + t * 16 + nn;
        int which = n >> 10;            // 0=Q 1=K 2=V
        int c     = n & 1023;
        int hh    = c >> 6, dd = c & 63;
        _Float16* dst = (which == 0) ? Qh : (which == 1) ? Kh : Vh;
        float scale   = (which == 0) ? 0.125f : 1.0f;   // 1/sqrt(64) folded into Q
        float bv = bias[n];
#pragma unroll
        for (int i = 0; i < 8; ++i) {
            int m = mbase + i;
            int bidx = m >> 11, trow = m & 2047;
            float val = (acc[t][i] + bv) * scale;
            dst[((size_t)(bidx * NH + hh) * T_SEQ + trow) * HD + dd] = (_Float16)val;
        }
    }
}

// ---------------------------------------------------------------------------
// Kernel 2: flash attention per (b,h,q-tile of 64). 4 waves, each owns 16 q
// rows. K tiles arrive in LDS via TDM (double buffered, overlapped one block
// ahead); S = Q K^T via WMMA; online softmax with shfl_xor reductions inside
// 16-lane halves; P re-laid out via LDS; O += P V with V transposed via LDS.
// ---------------------------------------------------------------------------
__global__ void __launch_bounds__(128)
attn_kernel(const _Float16* __restrict__ Qh, const _Float16* __restrict__ Kh,
            const _Float16* __restrict__ Vh, _Float16* __restrict__ Oh)
{
    __shared__ __align__(16) _Float16 Kt[2 * KT_SIZE];         // TDM K tiles
    __shared__ __align__(16) _Float16 Vt[64 * VT_STRIDE];      // V^T tile
    __shared__ __align__(16) _Float16 Ps[4 * 16 * P_STRIDE];   // per-wave P tiles

    const int tid  = threadIdx.x;
    const int wave = tid >> 5;
    const int lane = tid & 31;
    const int nn   = lane & 15;
    const int hs   = lane >> 4;
    const int qt   = blockIdx.x & 31;
    const int bh   = blockIdx.x >> 5;
    const size_t base = (size_t)bh * T_SEQ * HD;
    const int q0 = qt * 64;

    const _Float16* Kbase = Kh + base;
    const unsigned  kt_lds0 = (unsigned)(size_t)&Kt[0];
    const unsigned  kt_lds1 = (unsigned)(size_t)&Kt[KT_SIZE];

    // Prologue: start DMA of first K tile (wave 0 owns the TDM pipe)
    if (tid < 32) tdm_load_k_tile(Kbase, kt_lds0);

    // Q fragments (A layout): lane -> row (lane&15); K-dim 64 => 2 fragments
    const int qrow = q0 + wave * 16 + nn;
    v16h aQ[2];
#pragma unroll
    for (int s = 0; s < 2; ++s) {
        const _Float16* p = Qh + base + (size_t)qrow * HD + s * 32 + hs * 8;
        aQ[s] = make16(*(const v8h*)p, *(const v8h*)(p + 16));
    }

    v8f o[4];
#pragma unroll
    for (int t = 0; t < 4; ++t) o[t] = {};
    float rmax[8], rsum[8];
#pragma unroll
    for (int i = 0; i < 8; ++i) { rmax[i] = -3.0e38f; rsum[i] = 0.0f; }

    _Float16* Pw = Ps + wave * 16 * P_STRIDE;

    for (int kb = 0; kb < T_SEQ; kb += 64) {
        const int buf = (kb >> 6) & 1;
        // Current K tile DMA must be complete before anyone reads it; the
        // barrier also closes out last iteration's Vt/P/Kt[buf^1] reads.
        if (tid < 32) __builtin_amdgcn_s_wait_tensorcnt(0);
        __syncthreads();
        // Kick DMA for the next K tile into the other buffer (fully
        // overlapped with S/softmax/PV of this block).
        if (tid < 32 && (kb + 64 < T_SEQ))
            tdm_load_k_tile(Kbase + (size_t)(kb + 64) * HD, buf ? kt_lds0 : kt_lds1);

        // Stage V^T: Vt[d][k] so PV B-fragments are contiguous
#pragma unroll
        for (int it = 0; it < 8; ++it) {
            int slot = tid + it * 128;
            int kr = slot >> 4, ds = slot & 15;
            v4h vv = *(const v4h*)(Vh + base + (size_t)(kb + kr) * HD + ds * 4);
            Vt[(ds * 4 + 0) * VT_STRIDE + kr] = vv.x;
            Vt[(ds * 4 + 1) * VT_STRIDE + kr] = vv.y;
            Vt[(ds * 4 + 2) * VT_STRIDE + kr] = vv.z;
            Vt[(ds * 4 + 3) * VT_STRIDE + kr] = vv.w;
        }

        // S = Q K^T: B-column n is K row (kb + n), row-contiguous in Kt
        const _Float16* Kcur = Kt + buf * KT_SIZE;
        v8f s_acc[4];
#pragma unroll
        for (int t = 0; t < 4; ++t) {
            s_acc[t] = {};
            const _Float16* krow = Kcur + (t * 16 + nn) * KT_STRIDE + hs * 8;
            v16h bK0 = make16(*(const v8h*)(krow), *(const v8h*)(krow + 16));
            v16h bK1 = make16(*(const v8h*)(krow + 32), *(const v8h*)(krow + 48));
            s_acc[t] = wmma_f16(aQ[0], bK0, s_acc[t]);
            s_acc[t] = wmma_f16(aQ[1], bK1, s_acc[t]);
        }

        // Online softmax: row i lives on vgpr i across 16 lanes of this half
#pragma unroll
        for (int i = 0; i < 8; ++i) {
            float lm = fmaxf(fmaxf(s_acc[0][i], s_acc[1][i]),
                             fmaxf(s_acc[2][i], s_acc[3][i]));
#pragma unroll
            for (int m = 1; m < 16; m <<= 1) lm = fmaxf(lm, __shfl_xor(lm, m, 32));
            float Mn   = fmaxf(rmax[i], lm);
            float corr = __expf(rmax[i] - Mn);
            rmax[i] = Mn;
            float ps = 0.0f;
#pragma unroll
            for (int t = 0; t < 4; ++t) {
                float p = __expf(s_acc[t][i] - Mn);
                s_acc[t][i] = p;
                ps += p;
            }
#pragma unroll
            for (int m = 1; m < 16; m <<= 1) ps += __shfl_xor(ps, m, 32);
            rsum[i] = rsum[i] * corr + ps;
#pragma unroll
            for (int t = 0; t < 4; ++t) o[t][i] *= corr;
        }

        // Write P (C layout -> row-major LDS), then re-read as A fragments
#pragma unroll
        for (int t = 0; t < 4; ++t)
#pragma unroll
            for (int i = 0; i < 8; ++i)
                Pw[(i + hs * 8) * P_STRIDE + t * 16 + nn] = (_Float16)s_acc[t][i];
        __syncthreads();

        // O += P V
#pragma unroll
        for (int s = 0; s < 2; ++s) {
            const _Float16* prow = Pw + nn * P_STRIDE + s * 32 + hs * 8;
            v16h aP = make16(*(const v8h*)prow, *(const v8h*)(prow + 16));
            v16h bV[4];
#pragma unroll
            for (int t = 0; t < 4; ++t) {
                const _Float16* vcol = Vt + (t * 16 + nn) * VT_STRIDE + s * 32 + hs * 8;
                bV[t] = make16(*(const v8h*)vcol, *(const v8h*)(vcol + 16));
            }
#pragma unroll
            for (int t = 0; t < 4; ++t)
                o[t] = wmma_f16(aP, bV[t], o[t]);
        }
    }

    // Normalize and scatter to [B,T,C] f16
    const int b = bh >> 4, h = bh & 15;
#pragma unroll
    for (int t = 0; t < 4; ++t)
#pragma unroll
        for (int i = 0; i < 8; ++i) {
            int m = q0 + wave * 16 + hs * 8 + i;
            float val = o[t][i] / rsum[i];
            Oh[(size_t)(b * T_SEQ + m) * D_MODEL + h * HD + t * 16 + nn] = (_Float16)val;
        }
}

// ---------------------------------------------------------------------------
// Kernel 3: out = attn_out @ w_out + b_out  (A f16, B f32->f16, D f32)
// ---------------------------------------------------------------------------
__global__ void __launch_bounds__(256)
out_gemm_kernel(const _Float16* __restrict__ A, const float* __restrict__ W,
                const float* __restrict__ bias, float* __restrict__ out)
{
    __shared__ __align__(16) _Float16 As[128 * TA_STRIDE];
    __shared__ __align__(16) _Float16 Bs[64 * TB_STRIDE];

    const int tid  = threadIdx.x;
    const int wave = tid >> 5;
    const int lane = tid & 31;
    const int nn   = lane & 15;
    const int hs   = lane >> 4;
    const int row0 = blockIdx.x * 128;
    const int col0 = blockIdx.y * 64;

    v8f acc[4];
#pragma unroll
    for (int t = 0; t < 4; ++t) acc[t] = {};

    for (int kk = 0; kk < D_MODEL; kk += 32) {
#pragma unroll
        for (int it = 0; it < 4; ++it) {
            int slot = tid + it * 256;
            int r = slot >> 3, cs = slot & 7;
            v4h a = *(const v4h*)(A + (size_t)(row0 + r) * D_MODEL + kk + cs * 4);
            *(v4h*)(As + r * TA_STRIDE + cs * 4) = a;
        }
#pragma unroll
        for (int it = 0; it < 2; ++it) {
            int slot = tid + it * 256;
            int kr = slot >> 4, ns = slot & 15;
            v4f b = *(const v4f*)(W + (size_t)(kk + kr) * D_MODEL + col0 + ns * 4);
            Bs[(ns * 4 + 0) * TB_STRIDE + kr] = (_Float16)b.x;
            Bs[(ns * 4 + 1) * TB_STRIDE + kr] = (_Float16)b.y;
            Bs[(ns * 4 + 2) * TB_STRIDE + kr] = (_Float16)b.z;
            Bs[(ns * 4 + 3) * TB_STRIDE + kr] = (_Float16)b.w;
        }
        if (kk + 32 < D_MODEL) {
            __builtin_prefetch(A + (size_t)(row0 + (tid >> 3)) * D_MODEL + kk + 32 + (tid & 7) * 4, 0, 3);
            __builtin_prefetch(W + (size_t)(kk + 32 + (tid >> 4)) * D_MODEL + col0 + (tid & 15) * 4, 0, 3);
        }
        __syncthreads();

        const int kb = hs * 8;
        const _Float16* arow = As + (wave * 16 + nn) * TA_STRIDE;
        v16h afrag = make16(*(const v8h*)(arow + kb), *(const v8h*)(arow + kb + 16));
        v16h bfrag[4];
#pragma unroll
        for (int t = 0; t < 4; ++t) {
            const _Float16* bcol = Bs + (t * 16 + nn) * TB_STRIDE;
            bfrag[t] = make16(*(const v8h*)(bcol + kb), *(const v8h*)(bcol + kb + 16));
        }
#pragma unroll
        for (int t = 0; t < 4; ++t)
            acc[t] = wmma_f16(afrag, bfrag[t], acc[t]);
        __syncthreads();
    }

    const int mbase = row0 + wave * 16 + hs * 8;
#pragma unroll
    for (int t = 0; t < 4; ++t) {
        int n = col0 + t * 16 + nn;
        float bv = bias[n];
#pragma unroll
        for (int i = 0; i < 8; ++i) {
            int m = mbase + i;
            out[(size_t)m * D_MODEL + n] = acc[t][i] + bv;
        }
    }
}

// ---------------------------------------------------------------------------
extern "C" void kernel_launch(void* const* d_in, const int* in_sizes, int n_in,
                              void* d_out, int out_size, void* d_ws, size_t ws_size,
                              hipStream_t stream) {
    (void)in_sizes; (void)n_in; (void)out_size; (void)ws_size;
    const float* x     = (const float*)d_in[0];
    const float* w_qkv = (const float*)d_in[1];
    const float* b_qkv = (const float*)d_in[2];
    const float* w_out = (const float*)d_in[3];
    const float* b_out = (const float*)d_in[4];
    float* out = (float*)d_out;

    const size_t elems = (size_t)BATCH * NH * T_SEQ * HD;  // 4,194,304
    _Float16* Qh = (_Float16*)d_ws;
    _Float16* Kh = Qh + elems;
    _Float16* Vh = Kh + elems;
    _Float16* Oh = Vh + elems;   // [B,T,C] attention output, f16

    // qkv projection: M=4096 (32 tiles of 128), N=3072 (48 tiles of 64)
    qkv_gemm_kernel<<<dim3(32, 48), 256, 0, stream>>>(x, w_qkv, b_qkv, Qh, Kh, Vh);
    // flash attention: B*H*(T/64) = 2*16*32 blocks, 4 waves each
    attn_kernel<<<dim3(BATCH * NH * (T_SEQ / 64)), 128, 0, stream>>>(Qh, Kh, Vh, Oh);
    // output projection: M=4096, N=1024 (16 tiles of 64)
    out_gemm_kernel<<<dim3(32, 16), 256, 0, stream>>>(Oh, w_out, b_out, out);
}
